// GRIL_73847667687517
// MI455X (gfx1250) — matrix operand order repaired and checked by reading, not compiled, not run on hardware
//
#include <hip/hip_runtime.h>

// GRIL forward for MI455X (gfx1250, wave32).
// Graph-conv hops run as f16 WMMA GEMMs (v_wmma_f32_16x16x32_f16, f32 accum),
// with the adjacency column panel staged in LDS via the CDNA5 async
// global->LDS path (global_load_async_to_lds_b128 + s_wait_asynccnt), shared
// by all waves of the block; the X operand is software-pipelined.
// Channel mixes / gates in fp32 VALU.

typedef _Float16 half8 __attribute__((ext_vector_type(8)));
typedef _Float16 v16h  __attribute__((ext_vector_type(16)));
typedef float    v8f   __attribute__((ext_vector_type(8)));

#define NN 512
#define BB 16
#define HH 64
#define TT 64

// Padded LDS row stride for the B panel: 520 halfs = 1040 B.
// 1040 B = 260 dwords; 260 % 64 = 4 -> lane l hits bank group 4l (disjoint
// 4-bank spans across the 16 lanes of a half-wave) => conflict-free b128 reads.
#define BSTRIDE 520

#if __has_builtin(__builtin_amdgcn_global_load_async_to_lds_b128) && \
    __has_builtin(__builtin_amdgcn_s_wait_asynccnt)
#define USE_ASYNC_LDS 1
// Builtin prototype (from hipcc diagnostic): arg0 = int4 addrspace(1)* (global),
// arg1 = int4 addrspace(3)* (LDS), then imm offset, imm cpol.
typedef int v4i __attribute__((vector_size(16)));
typedef __attribute__((address_space(1))) v4i gv4i;
typedef __attribute__((address_space(3))) v4i sv4i;
#endif

__device__ __forceinline__ float sigm(float v) { return 1.f / (1.f + expf(-v)); }

// ---------------------------------------------------------------------------
// WMMA hop: Out[b, chOfs+ch, n] = sum_v X[b, ch, v] * Adj[n, v]   (X @ Adj^T)
// X: f16 [B, Cin, N] row-major. Adj: f16 [N, N] row-major (B-operand lane n
// holds Adj row n, contiguous in K).
// Block covers (wavesPerBlock) row tiles x 1 column tile; the column tile's
// full B panel (16 x 512 f16) is staged in LDS once and reused by all waves.
// Grid: (Mtiles/wavesPerBlock) * 32 blocks; colTile = blockIdx & 31.
// ---------------------------------------------------------------------------
template <typename OutT>
__global__ void hop_wmma(
    const _Float16* __restrict__ X, const _Float16* __restrict__ Adj,
    OutT* __restrict__ Out, int Cin, int Cout, int chOfs)
{
  __shared__ __align__(16) _Float16 sB[16 * BSTRIDE];

  const int wavesPerBlock = blockDim.x >> 5;
  const int waveId  = threadIdx.x >> 5;
  const int lane    = threadIdx.x & 31;
  const int l15     = lane & 15;
  const int colTile = blockIdx.x & 31;
  const int rowBlk  = blockIdx.x >> 5;
  const int n0      = colTile << 4;
  const int m0      = (rowBlk * wavesPerBlock + waveId) << 4;

  // ---- Stage B panel: Adj rows n0..n0+15, K = 0..511 (1024 x b128 chunks) --
#ifdef USE_ASYNC_LDS
  for (int c = threadIdx.x; c < 1024; c += blockDim.x) {
    int row = c >> 6, kc = c & 63;
    __builtin_amdgcn_global_load_async_to_lds_b128(
        (gv4i*)(Adj + (n0 + row) * NN + kc * 8),
        (sv4i*)&sB[row * BSTRIDE + kc * 8], 0, 0);
  }
  __builtin_amdgcn_s_wait_asynccnt(0);
  __syncthreads();
#else
  for (int c = threadIdx.x; c < 1024; c += blockDim.x) {
    int row = c >> 6, kc = c & 63;
    *(half8*)(&sB[row * BSTRIDE + kc * 8]) =
        *(const half8*)(Adj + (n0 + row) * NN + kc * 8);
  }
  __syncthreads();
#endif

  // A-operand (16x32 f16): lanes 0-15 hold row m, K{0..7,16..23};
  //                        lanes 16-31 hold row m, K{8..15,24..31}.
  const _Float16* Xrow = X + (m0 + l15) * NN + ((lane < 16) ? 0 : 8);
  // B-operand (32x16 f16): lanes 0-15 hold col n, K{0..7,16..23}; lanes 16-31 K{8..15,24..31}.
  const _Float16* Brow = &sB[l15 * BSTRIDE + ((lane < 16) ? 0 : 8)];

  v8f acc = {};
  // 2-stage pipeline on the global A fragments (B comes from LDS).
  half8 alo = *(const half8*)(Xrow);
  half8 ahi = *(const half8*)(Xrow + 16);
  for (int k0 = 0; k0 < NN; k0 += 32) {
    half8 blo = *(const half8*)(Brow + k0);
    half8 bhi = *(const half8*)(Brow + k0 + 16);
    v16h a, bm;
#pragma unroll
    for (int i = 0; i < 8; ++i) {
      a[i] = alo[i];  a[i + 8] = ahi[i];
      bm[i] = blo[i]; bm[i + 8] = bhi[i];
    }
    if (k0 + 32 < NN) {               // uniform branch: prefetch next A chunk
      alo = *(const half8*)(Xrow + k0 + 32);
      ahi = *(const half8*)(Xrow + k0 + 48);
    }
    acc = __builtin_amdgcn_wmma_f32_16x16x32_f16(
        /*neg_a=*/false, a, /*neg_b=*/false, bm,
        /*c_mod=*/(short)0, acc, /*reuse_a=*/false, /*reuse_b=*/false);
  }

  // C/D layout: VGPR j -> (M=j, N=lane) lanes<16 ; (M=j+8, N=lane-16) lanes>=16
  const int col = n0 + l15;
  const int rb  = m0 + ((lane < 16) ? 0 : 8);
#pragma unroll
  for (int j = 0; j < 8; ++j) {
    int r = rb + j;
    int bt = r / Cin, ch = r % Cin;
    Out[(bt * Cout + chOfs + ch) * NN + col] = (OutT)acc[j];
  }
}

// ---------------------------------------------------------------------------
// One-time setup: adj f32 -> f16 (both supports, 2*N*N = 524288 = B*H*N),
// broadcast h0 into h[b], write the scalar 0.0 output.
// ---------------------------------------------------------------------------
__global__ void k_setup(const float* __restrict__ adj, const float* __restrict__ h0,
                        _Float16* __restrict__ Ah, float* __restrict__ h,
                        float* __restrict__ scal)
{
  int i = blockIdx.x * blockDim.x + threadIdx.x;      // 524288 threads
  Ah[i] = (_Float16)adj[i];
  h[i]  = h0[i & (HH * NN - 1)];
  if (i == 0) *scal = 0.f;
}

// ---------------------------------------------------------------------------
// Step stage 1: xs_hat_1 (prediction), x1, x_in = W_li @ [x1, m, h]  -> f16
// One thread per (b, n) column.
// ---------------------------------------------------------------------------
__global__ void __launch_bounds__(256) k_pre(
    const float* __restrict__ x, const unsigned char* __restrict__ mask,
    const float* __restrict__ h,
    const float* __restrict__ Wfs, const float* __restrict__ bfs,
    const float* __restrict__ Wli, const float* __restrict__ bli,
    float* __restrict__ pred, _Float16* __restrict__ Xin, int t)
{
  int idx = blockIdx.x * blockDim.x + threadIdx.x;    // B*N = 8192
  int b = idx >> 9, n = idx & (NN - 1);

  float hc[HH];
#pragma unroll
  for (int c = 0; c < HH; ++c) hc[c] = h[(b * HH + c) * NN + n];

  float xs1 = bfs[0];
#pragma unroll
  for (int c = 0; c < HH; ++c) xs1 += Wfs[c] * hc[c];

  int xi = (b * NN + n) * TT + t;
  pred[xi] = xs1;
  bool m  = mask[xi] != 0;
  float x1 = m ? x[xi] : xs1;
  float mf = m ? 1.f : 0.f;

  float acc[HH];
#pragma unroll
  for (int o = 0; o < HH; ++o)
    acc[o] = bli[o] + Wli[o * 66] * x1 + Wli[o * 66 + 1] * mf;
  for (int c = 0; c < HH; ++c) {
    float v = hc[c];
#pragma unroll
    for (int o = 0; o < HH; ++o) acc[o] += Wli[o * 66 + 2 + c] * v;
  }
#pragma unroll
  for (int o = 0; o < HH; ++o) Xin[(b * HH + o) * NN + n] = (_Float16)acc[o];
}

// ---------------------------------------------------------------------------
// Step stage 2 (after decoder hops): W_gc, W_lo, PReLU, repr out, xs_hat_2,
// imputation out, x2, and Z0 = f16 [x2, m, h].
// ---------------------------------------------------------------------------
__global__ void __launch_bounds__(256) k_mid(
    const float* __restrict__ x, const unsigned char* __restrict__ mask,
    const float* __restrict__ h, const float* __restrict__ gc,
    const float* __restrict__ Wgc, const float* __restrict__ bgc,
    const float* __restrict__ Wlo, const float* __restrict__ blo,
    const float* __restrict__ Wro, const float* __restrict__ bro,
    const float* __restrict__ pa,
    float* __restrict__ repr, float* __restrict__ imp,
    _Float16* __restrict__ Z0, int t)
{
  int idx = blockIdx.x * blockDim.x + threadIdx.x;    // 8192
  int b = idx >> 9, n = idx & (NN - 1);

  float hc[HH];
#pragma unroll
  for (int c = 0; c < HH; ++c) hc[c] = h[(b * HH + c) * NN + n];

  float g[HH];
#pragma unroll
  for (int o = 0; o < HH; ++o) g[o] = bgc[o];
  for (int c = 0; c < 128; ++c) {
    float v = gc[(b * 128 + c) * NN + n];
#pragma unroll
    for (int o = 0; o < HH; ++o) g[o] += Wgc[o * 128 + c] * v;
  }

  float ov[HH];
#pragma unroll
  for (int o = 0; o < HH; ++o) ov[o] = blo[o];
  for (int c = 0; c < HH; ++c) {
    float v = g[c];
#pragma unroll
    for (int o = 0; o < HH; ++o) ov[o] += Wlo[o * 128 + c] * v;
  }
  for (int c = 0; c < HH; ++c) {
    float v = hc[c];
#pragma unroll
    for (int o = 0; o < HH; ++o) ov[o] += Wlo[o * 128 + 64 + c] * v;
  }

  float a = pa[0];
#pragma unroll
  for (int o = 0; o < HH; ++o) ov[o] = (ov[o] >= 0.f) ? ov[o] : a * ov[o];

#pragma unroll
  for (int o = 0; o < HH; ++o) {
    repr[((b * 128 + o) * NN + n) * TT + t]      = ov[o];
    repr[((b * 128 + 64 + o) * NN + n) * TT + t] = hc[o];
  }

  float xs2 = bro[0];
#pragma unroll
  for (int c = 0; c < HH; ++c) xs2 += Wro[c] * ov[c] + Wro[64 + c] * hc[c];

  int xi = (b * NN + n) * TT + t;
  imp[xi] = xs2;
  bool m  = mask[xi] != 0;
  float x2 = m ? x[xi] : xs2;
  float mf = m ? 1.f : 0.f;

  Z0[(b * 66 + 0) * NN + n] = (_Float16)x2;
  Z0[(b * 66 + 1) * NN + n] = (_Float16)mf;
#pragma unroll
  for (int o = 0; o < HH; ++o)
    Z0[(b * 66 + 2 + o) * NN + n] = (_Float16)hc[o];
}

// ---------------------------------------------------------------------------
// Gates: r, u = sigmoid(W @ xg); store u; Z0c = f16 [x2, m, r*h].
// Two threads per column (og splits the 64 outputs).
// ---------------------------------------------------------------------------
__global__ void __launch_bounds__(256) k_gates(
    const float* __restrict__ h,
    const _Float16* __restrict__ Z0, const _Float16* __restrict__ Z1a,
    const _Float16* __restrict__ Z2a, const _Float16* __restrict__ Z1b,
    const _Float16* __restrict__ Z2b,
    const float* __restrict__ Wr, const float* __restrict__ br,
    const float* __restrict__ Wu, const float* __restrict__ bu,
    float* __restrict__ ubuf, _Float16* __restrict__ Z0c)
{
  int idx = blockIdx.x * blockDim.x + threadIdx.x;    // 2*B*N = 16384
  int og = idx >> 13;
  int bn = idx & 8191;
  int b = bn >> 9, n = bn & (NN - 1);
  int ob = og * 32;

  float aR[32], aU[32];
#pragma unroll
  for (int o = 0; o < 32; ++o) { aR[o] = br[ob + o]; aU[o] = bu[ob + o]; }

  const _Float16* Zs[5] = {Z0, Z1a, Z2a, Z1b, Z2b};
#pragma unroll
  for (int blk = 0; blk < 5; ++blk) {
    const _Float16* Zp = Zs[blk] + (b * 66) * NN + n;
    int cb = blk * 66;
    for (int cc = 0; cc < 66; ++cc) {
      float v = (float)Zp[cc * NN];
      int c = cb + cc;
#pragma unroll
      for (int o = 0; o < 32; ++o) {
        aR[o] += Wr[(ob + o) * 330 + c] * v;
        aU[o] += Wu[(ob + o) * 330 + c] * v;
      }
    }
  }

#pragma unroll
  for (int o = 0; o < 32; ++o) {
    int oo = ob + o;
    float r  = sigm(aR[o]);
    float uu = sigm(aU[o]);
    int hi = (b * HH + oo) * NN + n;
    ubuf[hi] = uu;
    Z0c[(b * 66 + 2 + oo) * NN + n] = (_Float16)(r * h[hi]);
  }
  if (og == 0) {
    Z0c[(b * 66 + 0) * NN + n] = Z0[(b * 66 + 0) * NN + n];
    Z0c[(b * 66 + 1) * NN + n] = Z0[(b * 66 + 1) * NN + n];
  }
}

// ---------------------------------------------------------------------------
// Final: c = tanh(Wc @ xc); h_new = u*h + (1-u)*c; write states output.
// ---------------------------------------------------------------------------
__global__ void __launch_bounds__(256) k_final(
    const _Float16* __restrict__ Z0c, const _Float16* __restrict__ Z1a,
    const _Float16* __restrict__ Z2a, const _Float16* __restrict__ Z1b,
    const _Float16* __restrict__ Z2b,
    const float* __restrict__ Wc, const float* __restrict__ bc,
    const float* __restrict__ ubuf,
    float* __restrict__ h, float* __restrict__ states, int t)
{
  int idx = blockIdx.x * blockDim.x + threadIdx.x;    // 16384
  int og = idx >> 13;
  int bn = idx & 8191;
  int b = bn >> 9, n = bn & (NN - 1);
  int ob = og * 32;

  float ac[32];
#pragma unroll
  for (int o = 0; o < 32; ++o) ac[o] = bc[ob + o];

  const _Float16* Zs[5] = {Z0c, Z1a, Z2a, Z1b, Z2b};
#pragma unroll
  for (int blk = 0; blk < 5; ++blk) {
    const _Float16* Zp = Zs[blk] + (b * 66) * NN + n;
    int cb = blk * 66;
    for (int cc = 0; cc < 66; ++cc) {
      float v = (float)Zp[cc * NN];
      int c = cb + cc;
#pragma unroll
      for (int o = 0; o < 32; ++o) ac[o] += Wc[(ob + o) * 330 + c] * v;
    }
  }

#pragma unroll
  for (int o = 0; o < 32; ++o) {
    int oo = ob + o;
    float cg = tanhf(ac[o]);
    int hi = (b * HH + oo) * NN + n;
    float uu = ubuf[hi];
    float hn = uu * h[hi] + (1.f - uu) * cg;
    h[hi] = hn;
    states[((b * HH + oo) * NN + n) * TT + t] = hn;
  }
}

// ---------------------------------------------------------------------------
extern "C" void kernel_launch(void* const* d_in, const int* in_sizes, int n_in,
                              void* d_out, int out_size, void* d_ws, size_t ws_size,
                              hipStream_t stream)
{
  (void)in_sizes; (void)n_in; (void)out_size; (void)ws_size;

  const float* x    = (const float*)d_in[0];
  const unsigned char* mask = (const unsigned char*)d_in[1];  // numpy bool layout
  const float* adj  = (const float*)d_in[2];
  const float* h0   = (const float*)d_in[3];
  const float* Wfs  = (const float*)d_in[4];
  const float* bfs  = (const float*)d_in[5];
  const float* Wli  = (const float*)d_in[6];
  const float* bli  = (const float*)d_in[7];
  const float* Wgc  = (const float*)d_in[8];
  const float* bgc  = (const float*)d_in[9];
  const float* Wlo  = (const float*)d_in[10];
  const float* blo  = (const float*)d_in[11];
  const float* Wro  = (const float*)d_in[12];
  const float* bro  = (const float*)d_in[13];
  const float* pa   = (const float*)d_in[14];
  const float* Wr   = (const float*)d_in[15];
  const float* br   = (const float*)d_in[16];
  const float* Wu   = (const float*)d_in[17];
  const float* bu   = (const float*)d_in[18];
  const float* Wc   = (const float*)d_in[19];
  const float* bc   = (const float*)d_in[20];

  // Output slices (return order: imp, pred, repr, scalar, states), all f32.
  float* out    = (float*)d_out;
  float* imp    = out;                     // [16,1,512,64]   = 524288
  float* pred   = out + 524288;            // [16,1,512,64]   = 524288
  float* repr   = out + 1048576;           // [16,128,512,64] = 67108864
  float* scal   = out + 68157440;          // scalar 0.0
  float* states = out + 68157441;          // [1,16,64,512,64]= 33554432

  // Workspace carve-out (256B-aligned blocks, ~21 MB total).
  char* ws = (char*)d_ws;
  size_t ofs = 0;
  auto carve = [&](size_t bytes) -> char* {
    char* p = ws + ofs;
    ofs += (bytes + 255) & ~(size_t)255;
    return p;
  };
  _Float16* Ah  = (_Float16*)carve((size_t)2 * NN * NN * 2);      // A0h | A1h
  float*    h   = (float*)   carve((size_t)BB * HH * NN * 4);     // hidden state
  _Float16* Xin = (_Float16*)carve((size_t)BB * HH * NN * 2);     // decoder x_in
  float*    gc  = (float*)   carve((size_t)BB * 128 * NN * 4);    // decoder hops
  _Float16* Z[10];
  for (int i = 0; i < 10; ++i) Z[i] = (_Float16*)carve((size_t)BB * 66 * NN * 2);
  float*    ubuf = (float*)  carve((size_t)BB * HH * NN * 4);     // update gate
  _Float16* A0h = Ah;
  _Float16* A1h = Ah + (size_t)NN * NN;

  // One-time: convert adjacency to f16, broadcast h0, write scalar output.
  k_setup<<<2048, 256, 0, stream>>>(adj, h0, Ah, h, scal);

  // Hop grids: decoder M=1024 (64 row tiles, 8 waves/block, 256 thr);
  //            Z hops  M=1056 (66 row tiles, 6 waves/block, 192 thr).
  const dim3 gDec(8 * 32), bDec(256);
  const dim3 gZ(11 * 32),  bZ(192);

  for (int t = 0; t < TT; ++t) {
    // 1) prediction + decoder input
    k_pre<<<32, 256, 0, stream>>>(x, mask, h, Wfs, bfs, Wli, bli, pred, Xin, t);
    // 2) decoder graph conv (order 1, two supports) -> gc[:,0:64 | 64:128,:]
    hop_wmma<float><<<gDec, bDec, 0, stream>>>(Xin, A0h, gc, HH, 128, 0);
    hop_wmma<float><<<gDec, bDec, 0, stream>>>(Xin, A1h, gc, HH, 128, 64);
    // 3) decoder mixes, outputs, Z0 = [x2, m, h]
    k_mid<<<32, 256, 0, stream>>>(x, mask, h, gc, Wgc, bgc, Wlo, blo,
                                  Wro, bro, pa, repr, imp, Z[0], t);
    // 4) GRU gate gconv order-2: Z1a/Z1b then Z2a/Z2b
    hop_wmma<_Float16><<<gZ, bZ, 0, stream>>>(Z[0], A0h, Z[1], 66, 66, 0);
    hop_wmma<_Float16><<<gZ, bZ, 0, stream>>>(Z[0], A1h, Z[3], 66, 66, 0);
    hop_wmma<_Float16><<<gZ, bZ, 0, stream>>>(Z[1], A0h, Z[2], 66, 66, 0);
    hop_wmma<_Float16><<<gZ, bZ, 0, stream>>>(Z[3], A1h, Z[4], 66, 66, 0);
    // 5) r,u gates; Z0c = [x2, m, r*h]
    k_gates<<<64, 256, 0, stream>>>(h, Z[0], Z[1], Z[2], Z[3], Z[4],
                                    Wr, br, Wu, bu, ubuf, Z[5]);
    // 6) candidate gconv order-2
    hop_wmma<_Float16><<<gZ, bZ, 0, stream>>>(Z[5], A0h, Z[6], 66, 66, 0);
    hop_wmma<_Float16><<<gZ, bZ, 0, stream>>>(Z[5], A1h, Z[8], 66, 66, 0);
    hop_wmma<_Float16><<<gZ, bZ, 0, stream>>>(Z[6], A0h, Z[7], 66, 66, 0);
    hop_wmma<_Float16><<<gZ, bZ, 0, stream>>>(Z[8], A1h, Z[9], 66, 66, 0);
    // 7) candidate + hidden update + states output
    k_final<<<64, 256, 0, stream>>>(Z[5], Z[6], Z[7], Z[8], Z[9],
                                    Wc, bc, ubuf, h, states, t);
  }
}
